// CACE_18648747999780
// MI455X (gfx1250) — matrix operand order repaired and checked by reading, not compiled, not run on hardware
//
#include <hip/hip_runtime.h>
#include <hip/hip_bf16.h>
#include <cstdint>

// ---- CDNA5 WMMA vector types ----
typedef __attribute__((ext_vector_type(16))) _Float16 v16h;
typedef __attribute__((ext_vector_type(8)))  float    v8f;

// ---- problem constants (match reference) ----
#define N_NODES   10000
#define N_EDGES   60000
#define NRAD      8
#define DRE       8
#define DEC       9
#define NANG      40
#define PER_NODE  (NRAD*NANG*DEC)   // 2880
#define N_IN      504
#define KPAD      512
#define NHID      64
#define NGRAPH    4
#define RCUT      5.0f
#define PI_F      3.14159265358979f

// ---- workspace layout (bytes) ----
#define CNT_OFF   ((size_t)0)                     // int[10000]
#define CUR_OFF   ((size_t)40000)                 // int[10000]
#define OFS_OFF   ((size_t)80000)                 // int[10000]
#define EORD_OFF  ((size_t)120000)                // int[60000]
#define XB_OFF    ((size_t)524288)                // f16[10000*512]
#define BF_OFF    (XB_OFF + (size_t)N_NODES*KPAD*2)  // f16 W1 fragments (64KB)

__device__ __forceinline__ float sel3(float a, float b, float c, int i) {
    return (i == 0) ? a : ((i == 1) ? b : c);
}

// ---------------------------------------------------------------------------
// K0: zero per-node counters/cursors and the 4-graph energy output
// ---------------------------------------------------------------------------
__global__ void k_zero(int* __restrict__ cnt, int* __restrict__ cur,
                       float* __restrict__ out) {
    int i = blockIdx.x * blockDim.x + threadIdx.x;
    if (i < N_NODES) { cnt[i] = 0; cur[i] = 0; }
    if (i < NGRAPH) out[i] = 0.f;
}

// ---------------------------------------------------------------------------
// K1a: histogram edges by receiver node
// ---------------------------------------------------------------------------
__global__ void k_count(const int* __restrict__ eidx, int* __restrict__ cnt) {
    int e = blockIdx.x * blockDim.x + threadIdx.x;
    if (e < N_EDGES) atomicAdd(&cnt[eidx[N_EDGES + e]], 1);
}

// ---------------------------------------------------------------------------
// K1b: exclusive prefix-sum of counts (single block, chunked Hillis-Steele)
// ---------------------------------------------------------------------------
__global__ void __launch_bounds__(1024) k_scan(const int* __restrict__ cnt,
                                               int* __restrict__ offs) {
    __shared__ int sh[1024];
    __shared__ int sbase;
    if (threadIdx.x == 0) sbase = 0;
    __syncthreads();
    for (int c0 = 0; c0 < N_NODES; c0 += 1024) {
        int base = sbase;
        int i = c0 + (int)threadIdx.x;
        int v = (i < N_NODES) ? cnt[i] : 0;
        sh[threadIdx.x] = v;
        __syncthreads();
        for (int off = 1; off < 1024; off <<= 1) {
            int t = (threadIdx.x >= (unsigned)off) ? sh[threadIdx.x - off] : 0;
            __syncthreads();
            sh[threadIdx.x] += t;
            __syncthreads();
        }
        if (i < N_NODES) offs[i] = base + sh[threadIdx.x] - v;   // exclusive
        __syncthreads();
        if (threadIdx.x == 1023) sbase = base + sh[1023];
        __syncthreads();
    }
}

// ---------------------------------------------------------------------------
// K1c: scatter edge ids into per-node bins
// ---------------------------------------------------------------------------
__global__ void k_fill(const int* __restrict__ eidx, const int* __restrict__ offs,
                       int* __restrict__ cur, int* __restrict__ eord) {
    int e = blockIdx.x * blockDim.x + threadIdx.x;
    if (e >= N_EDGES) return;
    int nr = eidx[N_EDGES + e];
    int p = atomicAdd(&cur[nr], 1);
    eord[offs[nr] + p] = e;
}

// ---------------------------------------------------------------------------
// K2: swizzle W1[504,64] f32 -> f16 B-fragments, zero-padded K to 512.
// Layout: Bfrag[s][t][lane][e], lane holds column 16t+(lane&15),
// K = 32s + (lane<16 ? 0 : 16) + e  (contiguous K order, same as the A loads in k_gemm).
// ---------------------------------------------------------------------------
__global__ void k_w1(const float* __restrict__ W1, _Float16* __restrict__ Bfrag) {
    int tid = blockIdx.x * blockDim.x + threadIdx.x;
    if (tid >= 16 * 4 * 32 * 16) return;
    int e = tid & 15;
    int l = (tid >> 4) & 31;
    int t = (tid >> 9) & 3;
    int s = tid >> 11;
    int k   = 32 * s + ((l < 16) ? 0 : 16) + e;
    int col = 16 * t + (l & 15);
    float v = (k < N_IN) ? W1[k * NHID + col] : 0.f;
    Bfrag[tid] = (_Float16)v;
}

// ---------------------------------------------------------------------------
// K3: FUSED per-node kernel. One wave per node:
//   phase 1: walk this node's edges, accumulate A[8][360] in registers
//            (lane owns j = slot*32+lane for each radial index rr);
//            per edge only 12 ang*enc values + 96 FMAs.
//   phase 2: registers -> LDS (re-shard), radial transform + angular
//            contractions per (m,c) pair, emit f16 feature row.
// node_A never touches HBM.
// ---------------------------------------------------------------------------
__global__ void __launch_bounds__(128)
k_node(const int* __restrict__ offs, const int* __restrict__ cnt,
       const int* __restrict__ eord,
       const float* __restrict__ pos, const float* __restrict__ shifts,
       const int* __restrict__ species, const int* __restrict__ eidx,
       const float* __restrict__ Wsend, const float* __restrict__ Wrecv,
       const float* __restrict__ rtw, _Float16* __restrict__ Xb) {
    __shared__ float lA[4][PER_NODE];
    int wid = threadIdx.x >> 5, lane = threadIdx.x & 31;
    int n = blockIdx.x * 4 + wid;
    bool active = (n < N_NODES);

    float acc[8][12];
#pragma unroll
    for (int rr = 0; rr < 8; ++rr)
#pragma unroll
        for (int s = 0; s < 12; ++s) acc[rr][s] = 0.f;

    if (active) {
        int e0 = offs[n];
        int ne = cnt[n];
        float rx = pos[3*n+0], ry = pos[3*n+1], rz = pos[3*n+2];
        int zr = species[n];
        float er0 = Wrecv[zr*3+0], er1 = Wrecv[zr*3+1], er2 = Wrecv[zr*3+2];

        for (int k = 0; k < ne; ++k) {
            int e  = eord[e0 + k];
            int ns = eidx[e];
            float dx = rx - pos[3*ns+0] + shifts[3*e+0];
            float dy = ry - pos[3*ns+1] + shifts[3*e+1];
            float dz = rz - pos[3*ns+2] + shifts[3*e+2];
            float rl = sqrtf(dx*dx + dy*dy + dz*dz);
            float inv = 1.f / (rl + 1e-12f);
            float v0 = dx*inv, v1 = dy*inv, v2 = dz*inv;
            float cut = (rl < RCUT) ? 0.5f * (__cosf(PI_F * rl / RCUT) + 1.f) : 0.f;
            float coef = sqrtf(2.f / RCUT) * inv * cut;
            float radc[NRAD];
#pragma unroll
            for (int q = 0; q < NRAD; ++q)
                radc[q] = coef * __sinf((float)(q + 1) * PI_F * rl / RCUT);
            int zs = species[ns];
            float es0 = Wsend[zs*3+0], es1 = Wsend[zs*3+1], es2 = Wsend[zs*3+2];

            // ae[slot] = ang(a(j)) * enc(c(j)) for this lane's j = slot*32+lane
            float ae[12];
#pragma unroll
            for (int s = 0; s < 12; ++s) {
                int j = s * 32 + lane;            // j = a*9 + c
                if (j < NANG * DEC) {
                    int a = j / 9, c = j - a * 9;
                    float ang;
                    if (a == 0)       ang = 1.f;
                    else if (a < 4)   ang = sel3(v0, v1, v2, a - 1);
                    else if (a < 13) { int t = a - 4;
                        ang = sel3(v0, v1, v2, t / 3) * sel3(v0, v1, v2, t % 3); }
                    else             { int t = a - 13;
                        ang = sel3(v0, v1, v2, t / 9) * sel3(v0, v1, v2, (t / 3) % 3)
                            * sel3(v0, v1, v2, t % 3); }
                    float enc = sel3(es0, es1, es2, c / 3) * sel3(er0, er1, er2, c % 3);
                    ae[s] = ang * enc;
                } else {
                    ae[s] = 0.f;
                }
            }
#pragma unroll
            for (int rr = 0; rr < NRAD; ++rr)
#pragma unroll
                for (int s = 0; s < 12; ++s)
                    acc[rr][s] += radc[rr] * ae[s];
        }
        // re-shard through LDS for the contraction phase
#pragma unroll
        for (int rr = 0; rr < NRAD; ++rr)
#pragma unroll
            for (int s = 0; s < 12; ++s) {
                int j = s * 32 + lane;
                if (j < NANG * DEC) lA[wid][rr * 360 + j] = acc[rr][s];
            }
    }
    __syncthreads();
    if (!active) return;

    _Float16* xrow = Xb + (size_t)n * KPAD;
    if (lane < KPAD - N_IN) xrow[N_IN + lane] = (_Float16)0.f;   // pad 504..511

    const int BEG[4] = {0, 1, 4, 13};
    const int END[4] = {1, 4, 13, 40};
    for (int p = lane; p < DRE * DEC; p += 32) {                  // (m,c) pairs
        int m = p / 9, c = p - (p / 9) * 9;
        float wl[4][NRAD];
#pragma unroll
        for (int l = 0; l < 4; ++l)
#pragma unroll
            for (int r = 0; r < NRAD; ++r)
                wl[l][r] = rtw[((l * NRAD + r) * DRE + m) * DEC + c];
        float Tv[NANG];
#pragma unroll
        for (int l = 0; l < 4; ++l) {
#pragma unroll
            for (int a = BEG[l]; a < END[l]; ++a) {
                float t = 0.f;
#pragma unroll
                for (int r = 0; r < NRAD; ++r)
                    t += lA[wid][r * 360 + a * 9 + c] * wl[l][r];
                Tv[a] = t;
            }
        }
        float S[7];
        S[0] = Tv[0];
        S[1] = Tv[1]*Tv[1] + Tv[2]*Tv[2] + Tv[3]*Tv[3];
        { float s2 = 0.f;
#pragma unroll
          for (int a = 4; a < 13; ++a) s2 += Tv[a]*Tv[a];  S[2] = s2; }
        { float s3 = 0.f;
#pragma unroll
          for (int a = 13; a < 40; ++a) s3 += Tv[a]*Tv[a]; S[3] = s3; }
        { float s = 0.f;                                   // (l0,l1)=(1,1)
#pragma unroll
          for (int i = 0; i < 3; ++i)
#pragma unroll
            for (int j = 0; j < 3; ++j)
              s += Tv[1+i] * Tv[4 + 3*i + j] * Tv[1+j];
          S[4] = s; }
        { float s = 0.f;                                   // (1,2)
#pragma unroll
          for (int i = 0; i < 3; ++i)
#pragma unroll
            for (int j = 0; j < 9; ++j)
              s += Tv[1+i] * Tv[13 + 9*i + j] * Tv[4+j];
          S[5] = s; }
        { float s = 0.f;                                   // (2,1)
#pragma unroll
          for (int i = 0; i < 9; ++i)
#pragma unroll
            for (int j = 0; j < 3; ++j)
              s += Tv[4+i] * Tv[13 + 3*i + j] * Tv[1+j];
          S[6] = s; }
#pragma unroll
        for (int k = 0; k < 7; ++k)
            xrow[(m * 7 + k) * DEC + c] = (_Float16)S[k];
    }
}

// ---------------------------------------------------------------------------
// K4: ANN head.  X[10000,512]f16 @ W1frag -> v_wmma_f32_16x16x32_f16, then
// fused bias + silu + W2 matvec + per-graph atomic segment-sum.
// One wave per 16-row tile; 4 accumulators cover all 64 hidden columns.
// ---------------------------------------------------------------------------
union FragU { uint4 q[2]; v16h v; };

__global__ void __launch_bounds__(128)
k_gemm(const _Float16* __restrict__ Xb, const _Float16* __restrict__ Bfrag,
       const float* __restrict__ b1, const float* __restrict__ W2,
       const float* __restrict__ b2, const int* __restrict__ batch,
       float* __restrict__ out) {
    int wid = threadIdx.x >> 5, lane = threadIdx.x & 31;
    int mt = blockIdx.x * 4 + wid;
    if (mt >= N_NODES / 16) return;            // 625 tiles, wave-uniform exit
    int row0 = mt * 16;
    int cl = lane & 15, hi = lane >> 4;

    v8f acc[4];
#pragma unroll
    for (int t = 0; t < 4; ++t) acc[t] = (v8f){0.f,0.f,0.f,0.f,0.f,0.f,0.f,0.f};

    // A row pointer, contiguous K order matching Bfrag: lanes 0-15 K{0..15},
    // lanes 16-31 K{16..31}
    const _Float16* arow = Xb + (size_t)(row0 + cl) * KPAD + (hi ? 16 : 0);
    const _Float16* bbase = Bfrag + (size_t)lane * 16;

    for (int s = 0; s < KPAD / 32; ++s) {
        FragU af;
        af.q[0] = *(const uint4*)(arow + s * 32);
        af.q[1] = *(const uint4*)(arow + s * 32 + 8);
        if (s + 1 < KPAD / 32)
            __builtin_prefetch(arow + (s + 1) * 32, 0, 0);  // global_prefetch_b8
#pragma unroll
        for (int t = 0; t < 4; ++t) {
            FragU bf;
            const _Float16* bp = bbase + ((size_t)(s * 4 + t)) * 512;
            bf.q[0] = *(const uint4*)(bp);
            bf.q[1] = *(const uint4*)(bp + 8);
            acc[t] = __builtin_amdgcn_wmma_f32_16x16x32_f16(
                false, af.v, false, bf.v, (short)0, acc[t], false, false);
        }
    }

    // Epilogue: lane holds h_pre[row0 + j + 8*hi][16t + cl] in acc[t][j]
    float rowsum[8];
#pragma unroll
    for (int j = 0; j < 8; ++j) rowsum[j] = 0.f;
#pragma unroll
    for (int t = 0; t < 4; ++t) {
        int col = t * 16 + cl;
        float bb = b1[col];
        float w2 = W2[col];
#pragma unroll
        for (int j = 0; j < 8; ++j) {
            float x = acc[t][j] + bb;
            float h = x / (1.f + __expf(-x));              // silu
            rowsum[j] += h * w2;
        }
    }
    // butterfly-reduce across the 16 lanes of each half-wave (wave32)
#pragma unroll
    for (int j = 0; j < 8; ++j)
#pragma unroll
        for (int off = 1; off < 16; off <<= 1)
            rowsum[j] += __shfl_xor(rowsum[j], off, 32);

    if (cl == 0) {
        float bias2 = b2[0];
#pragma unroll
        for (int j = 0; j < 8; ++j) {
            int rown = row0 + hi * 8 + j;
            atomicAdd(&out[batch[rown]], rowsum[j] + bias2);
        }
    }
}

// ---------------------------------------------------------------------------
extern "C" void kernel_launch(void* const* d_in, const int* in_sizes, int n_in,
                              void* d_out, int out_size, void* d_ws, size_t ws_size,
                              hipStream_t stream) {
    const float* positions = (const float*)d_in[0];
    const float* shifts    = (const float*)d_in[1];
    const int*   species   = (const int*)  d_in[2];
    const int*   eidx      = (const int*)  d_in[3];
    const int*   batch     = (const int*)  d_in[4];
    const float* W_send    = (const float*)d_in[5];
    const float* W_recv    = (const float*)d_in[6];
    const float* rtw       = (const float*)d_in[7];
    const float* W1        = (const float*)d_in[8];
    const float* b1        = (const float*)d_in[9];
    const float* W2        = (const float*)d_in[10];
    const float* b2        = (const float*)d_in[11];
    float* out = (float*)d_out;

    char* ws = (char*)d_ws;
    int*       cnt   = (int*)(ws + CNT_OFF);
    int*       cur   = (int*)(ws + CUR_OFF);
    int*       offs  = (int*)(ws + OFS_OFF);
    int*       eord  = (int*)(ws + EORD_OFF);
    _Float16*  Xb    = (_Float16*)(ws + XB_OFF);
    _Float16*  Bfrag = (_Float16*)(ws + BF_OFF);

    k_zero <<<(N_NODES + 255) / 256, 256, 0, stream>>>(cnt, cur, out);
    k_count<<<(N_EDGES + 255) / 256, 256, 0, stream>>>(eidx, cnt);
    k_scan <<<1, 1024, 0, stream>>>(cnt, offs);
    k_fill <<<(N_EDGES + 255) / 256, 256, 0, stream>>>(eidx, offs, cur, eord);
    k_w1   <<<(16 * 4 * 32 * 16 + 255) / 256, 256, 0, stream>>>(W1, Bfrag);
    k_node <<<(N_NODES + 3) / 4, 128, 0, stream>>>(
        offs, cnt, eord, positions, shifts, species, eidx,
        W_send, W_recv, rtw, Xb);
    k_gemm <<<(N_NODES / 16 + 3) / 4, 128, 0, stream>>>(
        Xb, Bfrag, b1, W2, b2, batch, out);
}
